// EvolveGCN_15135464751705
// MI455X (gfx1250) — matrix operand order repaired and checked by reading, not compiled, not run on hardware
//
#include <hip/hip_runtime.h>
#include <hip/hip_bf16.h>

// ---- problem constants (from reference) ----
constexpr int N  = 50000;
constexpr int D  = 256;     // = k for top-k
constexpr int E  = 800000;

typedef __attribute__((ext_vector_type(16))) __bf16 v16bf;
typedef __attribute__((ext_vector_type(8)))  float  v8f;

// fp32 -> order-preserving uint key (larger float => larger key)
__device__ __forceinline__ unsigned sortable(float f) {
    unsigned u = __float_as_uint(f);
    return (u & 0x80000000u) ? ~u : (u | 0x80000000u);
}

// ---------------------------------------------------------------------------
// 0) zero-init: d_out (accumulated by scatter), deg, hist, counters
// ---------------------------------------------------------------------------
__global__ void k_init(float* __restrict__ out, float* __restrict__ deg,
                       unsigned* __restrict__ hist, unsigned* __restrict__ cnt) {
    int stride = gridDim.x * blockDim.x;
    int t0 = blockIdx.x * blockDim.x + threadIdx.x;
    for (int i = t0; i < N * D; i += stride) out[i] = 0.0f;
    for (int i = t0; i < N;     i += stride) deg[i] = 0.0f;
    for (int i = t0; i < 2048;  i += stride) hist[i] = 0u;
    for (int i = t0; i < 8;     i += stride) cnt[i] = 0u;
}

// ---------------------------------------------------------------------------
// 1) 1/||p||  (single block, LDS reduction)
// ---------------------------------------------------------------------------
__global__ void k_pnorm(const float* __restrict__ p, float* __restrict__ pn) {
    __shared__ float red[256];
    float v = p[threadIdx.x];
    red[threadIdx.x] = v * v;
    __syncthreads();
    for (int s = 128; s > 0; s >>= 1) {
        if (threadIdx.x < s) red[threadIdx.x] += red[threadIdx.x + s];
        __syncthreads();
    }
    if (threadIdx.x == 0) pn[0] = rsqrtf(red[0]);
}

// ---------------------------------------------------------------------------
// 2) score[i] = dot(X[i], p) * (1/||p||)   — one wave32 per row, 8 elts/lane
// ---------------------------------------------------------------------------
__global__ void k_score(const float* __restrict__ X, const float* __restrict__ p,
                        const float* __restrict__ pn, float* __restrict__ score) {
    int wv   = (blockIdx.x * blockDim.x + threadIdx.x) >> 5;
    int lane = threadIdx.x & 31;
    if (wv >= N) return;
    const float* xr = X + (size_t)wv * D + lane * 8;
    const float* pr = p + lane * 8;
    float4 a = *(const float4*)(xr),     b = *(const float4*)(xr + 4);
    float4 c = *(const float4*)(pr),     d = *(const float4*)(pr + 4);
    float s = a.x*c.x + a.y*c.y + a.z*c.z + a.w*c.w
            + b.x*d.x + b.y*d.y + b.z*d.z + b.w*d.w;
    for (int o = 16; o > 0; o >>= 1) s += __shfl_xor(s, o, 32);
    if (lane == 0) score[wv] = s * pn[0];
}

// ---------------------------------------------------------------------------
// 3) top-k(256) via radix histogram on key[31:21] (2048 bins)
// ---------------------------------------------------------------------------
__global__ void k_hist(const float* __restrict__ score, unsigned* __restrict__ hist) {
    int stride = gridDim.x * blockDim.x;
    for (int i = blockIdx.x * blockDim.x + threadIdx.x; i < N; i += stride) {
        unsigned bin = sortable(score[i]) >> 21;
        atomicAdd(&hist[bin], 1u);
    }
}

__global__ void k_thresh(const unsigned* __restrict__ hist, unsigned* __restrict__ cnt) {
    if (blockIdx.x || threadIdx.x) return;
    unsigned cum = 0;
    for (int b = 2047; b >= 0; --b) {
        unsigned h = hist[b];
        if (cum + h >= 256u) { cnt[2] = (unsigned)b; cnt[3] = 256u - cum; return; }
        cum += h;
    }
    cnt[2] = 0u; cnt[3] = 256u - cum;   // fallback (never hit for N>=256)
}

__global__ void k_collect(const float* __restrict__ score, unsigned* __restrict__ cnt,
                          unsigned* __restrict__ keys_sel, int* __restrict__ idx_sel) {
    unsigned B = cnt[2], R = cnt[3];
    int stride = gridDim.x * blockDim.x;
    for (int i = blockIdx.x * blockDim.x + threadIdx.x; i < N; i += stride) {
        unsigned u = sortable(score[i]);
        unsigned bin = u >> 21;
        if (bin > B) {
            unsigned s = atomicAdd(&cnt[0], 1u);
            keys_sel[s] = u; idx_sel[s] = i;
        } else if (bin == B) {
            unsigned t = atomicAdd(&cnt[1], 1u);
            if (t < R) {
                unsigned s = atomicAdd(&cnt[0], 1u);
                keys_sel[s] = u; idx_sel[s] = i;
            }
        }
    }
}

// single-block bitonic sort, descending by key; emits sorted node indices
__global__ void k_sort(const unsigned* __restrict__ keys_sel, const int* __restrict__ idx_sel,
                       int* __restrict__ idx_sorted) {
    __shared__ unsigned sk[256];
    __shared__ int      si[256];
    unsigned t = threadIdx.x;
    sk[t] = keys_sel[t]; si[t] = idx_sel[t];
    for (unsigned k = 2; k <= 256; k <<= 1) {
        for (unsigned j = k >> 1; j > 0; j >>= 1) {
            __syncthreads();
            unsigned i = t ^ j;
            if (i > t) {
                bool desc = ((t & k) == 0);
                unsigned a = sk[t], b = sk[i];
                if (desc ? (a < b) : (a > b)) {
                    sk[t] = b; sk[i] = a;
                    int ti = si[t]; si[t] = si[i]; si[i] = ti;
                }
            }
        }
    }
    __syncthreads();
    idx_sorted[t] = si[t];
}

// ---------------------------------------------------------------------------
// 4) x_tilde[r][c] = X[idx[r]][c] * tanh(score[idx[r]])
// ---------------------------------------------------------------------------
__global__ void k_xtilde(const float* __restrict__ X, const float* __restrict__ score,
                         const int* __restrict__ idx_sorted, float* __restrict__ xt) {
    int r = blockIdx.x, c = threadIdx.x;
    int idx = idx_sorted[r];
    float g = tanhf(score[idx]);
    xt[(size_t)r * D + c] = X[(size_t)idx * D + c] * g;
}

// ---------------------------------------------------------------------------
// 5) single-step GRU -> W_new, stored transposed in bf16 (WMMA B layout wants
//    contiguous K per output column). LDS stages the two row vectors.
// ---------------------------------------------------------------------------
__global__ void k_gru(const float* __restrict__ xt,  const float* __restrict__ W0,
                      const float* __restrict__ Wih, const float* __restrict__ Whh,
                      const float* __restrict__ bih, const float* __restrict__ bhh,
                      __bf16* __restrict__ WnT) {
    int r = blockIdx.x, c = threadIdx.x;
    __shared__ float sx[D], sh[D];
    sx[c] = xt[(size_t)r * D + c];
    sh[c] = W0[(size_t)r * D + c];
    __syncthreads();
    const float* wir = Wih + (size_t)c * D;
    const float* wiz = wir + (size_t)D * D;
    const float* win = wiz + (size_t)D * D;
    const float* whr = Whh + (size_t)c * D;
    const float* whz = whr + (size_t)D * D;
    const float* whn = whz + (size_t)D * D;
    float ir = 0.f, iz = 0.f, inn = 0.f, hr = 0.f, hz = 0.f, hn = 0.f;
    for (int k = 0; k < D; ++k) {
        float xv = sx[k], hv = sh[k];
        ir += xv * wir[k]; iz += xv * wiz[k]; inn += xv * win[k];
        hr += hv * whr[k]; hz += hv * whz[k]; hn  += hv * whn[k];
    }
    ir += bih[c];       hr += bhh[c];
    iz += bih[c + D];   hz += bhh[c + D];
    inn += bih[c + 2*D]; hn += bhh[c + 2*D];
    float rg = 1.f / (1.f + __expf(-(ir + hr)));
    float zg = 1.f / (1.f + __expf(-(iz + hz)));
    float ng = tanhf(inn + rg * hn);
    float wnew = (1.f - zg) * ng + zg * sh[c];
    WnT[(size_t)c * D + r] = (__bf16)wnew;   // transposed store
}

// ---------------------------------------------------------------------------
// 6) xw = X @ W_new  via V_WMMA_F32_16X16X32_BF16.
//    One wave per 16x16 tile; 8 WMMAs over K=256. A is converted fp32->bf16
//    on the fly; lane layout per ISA §7.12.2 (16-bit A 16x32, f32 C 16x16).
// ---------------------------------------------------------------------------
__global__ void k_gemm_wmma(const float* __restrict__ X, const __bf16* __restrict__ WnT,
                            float* __restrict__ xw) {
    int lane = threadIdx.x & 31;
    int wv   = threadIdx.x >> 5;
    int m0   = blockIdx.x * 16;
    int n0   = (blockIdx.y * 4 + wv) * 16;
    int half = lane >> 4;
    int l15  = lane & 15;
    const float*  xr = X   + (size_t)(m0 + l15) * D;   // A: row = M
    const __bf16* bc = WnT + (size_t)(n0 + l15) * D;   // B: row = N (transposed W_new)
    v8f c = {0.f, 0.f, 0.f, 0.f, 0.f, 0.f, 0.f, 0.f};
#pragma unroll
    for (int kb = 0; kb < D; kb += 32) {
        const float4* pa0 = (const float4*)(xr + kb + half * 8);
        const float4* pa1 = (const float4*)(xr + kb + 16 + half * 8);
        float4 f0 = pa0[0], f1 = pa0[1], f2 = pa1[0], f3 = pa1[1];
        v16bf a;
        a[0]  = (__bf16)f0.x; a[1]  = (__bf16)f0.y; a[2]  = (__bf16)f0.z; a[3]  = (__bf16)f0.w;
        a[4]  = (__bf16)f1.x; a[5]  = (__bf16)f1.y; a[6]  = (__bf16)f1.z; a[7]  = (__bf16)f1.w;
        a[8]  = (__bf16)f2.x; a[9]  = (__bf16)f2.y; a[10] = (__bf16)f2.z; a[11] = (__bf16)f2.w;
        a[12] = (__bf16)f3.x; a[13] = (__bf16)f3.y; a[14] = (__bf16)f3.z; a[15] = (__bf16)f3.w;
        typedef __attribute__((ext_vector_type(8))) __bf16 v8bf;
        v8bf b0 = *(const v8bf*)(bc + kb + half * 8);
        v8bf b1 = *(const v8bf*)(bc + kb + 16 + half * 8);
        v16bf b = __builtin_shufflevector(b0, b1, 0, 1, 2, 3, 4, 5, 6, 7, 8, 9, 10, 11, 12, 13, 14, 15);   // 16B-aligned, table-order K (matches A)
        c = __builtin_amdgcn_wmma_f32_16x16x32_bf16(
                /*neg_a=*/false, a, /*neg_b=*/false, b,
                /*c_mod=*/(short)0, c, /*reuse_a=*/false, /*reuse_b=*/false);
    }
    // C layout: VGPR r -> rows (m0 + half*8 + r), col = n0 + (lane&15)
    float* orow = xw + (size_t)(m0 + half * 8) * D + (n0 + l15);
#pragma unroll
    for (int r = 0; r < 8; ++r) orow[(size_t)r * D] = c[r];
}

// ---------------------------------------------------------------------------
// 7) degree (incl. self-loops), then dinv = rsqrt(deg)
// ---------------------------------------------------------------------------
__global__ void k_deg(const int* __restrict__ ei, const float* __restrict__ ew,
                      float* __restrict__ deg) {
    int t = blockIdx.x * blockDim.x + threadIdx.x;
    if (t < E)           atomicAdd(&deg[ei[E + t]], ew[t]);   // dst = row 1
    else if (t < E + N)  atomicAdd(&deg[t - E], 1.0f);        // self-loop
}

__global__ void k_dinv(const float* __restrict__ deg, float* __restrict__ dinv) {
    int t = blockIdx.x * blockDim.x + threadIdx.x;
    if (t < N) { float d = deg[t]; dinv[t] = (d > 0.f) ? rsqrtf(d) : 0.f; }
}

// ---------------------------------------------------------------------------
// 8) message scatter: one wave32 per (edge | self-loop), 8 f32 atomics/lane
// ---------------------------------------------------------------------------
__global__ void k_scatter(const int* __restrict__ ei, const float* __restrict__ ew,
                          const float* __restrict__ dinv, const float* __restrict__ xw,
                          float* __restrict__ out) {
    int gid  = blockIdx.x * blockDim.x + threadIdx.x;
    int wv   = gid >> 5;
    int lane = gid & 31;
    if (wv >= E + N) return;
    int src, dst; float w;
    if (wv < E) { src = ei[wv]; dst = ei[E + wv]; w = ew[wv]; }
    else        { src = dst = wv - E; w = 1.0f; }
    float norm = dinv[src] * w * dinv[dst];
    const float* xs = xw + (size_t)src * D;
    float*       od = out + (size_t)dst * D;
#pragma unroll
    for (int u = 0; u < 8; ++u) {
        int col = lane + u * 32;                 // coalesced across the wave
        atomicAdd(&od[col], norm * xs[col]);
    }
}

// ---------------------------------------------------------------------------
extern "C" void kernel_launch(void* const* d_in, const int* in_sizes, int n_in,
                              void* d_out, int out_size, void* d_ws, size_t ws_size,
                              hipStream_t stream) {
    const float* X   = (const float*)d_in[0];
    const int*   ei  = (const int*)  d_in[1];
    const float* ew  = (const float*)d_in[2];
    const float* p   = (const float*)d_in[3];
    const float* Wih = (const float*)d_in[4];
    const float* Whh = (const float*)d_in[5];
    const float* bih = (const float*)d_in[6];
    const float* bhh = (const float*)d_in[7];
    const float* W0  = (const float*)d_in[8];
    float* out = (float*)d_out;

    // ---- carve workspace (256B-aligned chunks) ----
    char* ws = (char*)d_ws;
    auto carve = [&](size_t bytes) -> char* {
        char* r = ws;
        ws += (bytes + 255) & ~size_t(255);
        return r;
    };
    float*    score      = (float*)   carve((size_t)N * 4);
    float*    pn         = (float*)   carve(16);
    unsigned* hist       = (unsigned*)carve(2048 * 4);
    unsigned* cnt        = (unsigned*)carve(8 * 4);
    unsigned* keys_sel   = (unsigned*)carve(256 * 4);
    int*      idx_sel    = (int*)     carve(256 * 4);
    int*      idx_sorted = (int*)     carve(256 * 4);
    float*    xt         = (float*)   carve((size_t)D * D * 4);
    __bf16*   WnT        = (__bf16*)  carve((size_t)D * D * 2);
    float*    deg        = (float*)   carve((size_t)N * 4);
    float*    dinv       = (float*)   carve((size_t)N * 4);
    float*    xw         = (float*)   carve((size_t)N * D * 4);
    (void)ws_size; (void)n_in; (void)in_sizes; (void)out_size;

    // 0) zero accumulators
    k_init<<<4096, 256, 0, stream>>>(out, deg, hist, cnt);
    // 1-2) scores
    k_pnorm<<<1, 256, 0, stream>>>(p, pn);
    k_score<<<(N * 32 + 255) / 256, 256, 0, stream>>>(X, p, pn, score);
    // 3) top-256 select + sort
    k_hist<<<512, 256, 0, stream>>>(score, hist);
    k_thresh<<<1, 1, 0, stream>>>(hist, cnt);
    k_collect<<<512, 256, 0, stream>>>(score, cnt, keys_sel, idx_sel);
    k_sort<<<1, 256, 0, stream>>>(keys_sel, idx_sel, idx_sorted);
    // 4) gated gather
    k_xtilde<<<D, D, 0, stream>>>(X, score, idx_sorted, xt);
    // 5) GRU -> W_new^T (bf16)
    k_gru<<<D, D, 0, stream>>>(xt, W0, Wih, Whh, bih, bhh, WnT);
    // 6) big GEMM via WMMA: 3125 x 4 blocks, 4 waves each = 50000 16x16 tiles
    k_gemm_wmma<<<dim3(N / 16, 4), 128, 0, stream>>>(X, WnT, xw);
    // 7) degrees + normalization
    k_deg<<<(E + N + 255) / 256, 256, 0, stream>>>(ei, ew, deg);
    k_dinv<<<(N + 255) / 256, 256, 0, stream>>>(deg, dinv);
    // 8) normalized scatter-add
    k_scatter<<<((size_t)(E + N) * 32 + 255) / 256, 256, 0, stream>>>(ei, ew, dinv, xw, out);
}
